// HierarchyVQmodulator_86912958202565
// MI455X (gfx1250) — compile-verified, matrix operand order checked by reading, not run on hardware
//
#include <hip/hip_runtime.h>
#include <hip/hip_bf16.h>
#include <math.h>

// Problem constants
#define B_ROWS   32768
#define NE       1024
#define ED       256
#define KL_W     5e-4f

typedef __attribute__((ext_vector_type(16))) __bf16 v16bf;
typedef __attribute__((ext_vector_type(8)))  __bf16 v8bf;
typedef __attribute__((ext_vector_type(8)))  float  v8f;

// d_out layout (float elements, reference tuple concatenated flat)
#define ZQ_OFF    0ull                       // 32768*256
#define LOSS_OFF  8388608ull                 // 1
#define SAMP_OFF  8388609ull                 // 32768*1024
#define IND_OFF   41943041ull                // 32768
#define CBV_OFF   41975809ull
#define MIND_OFF  41975810ull
#define HSW_OFF   41975811ull
#define CBL_OFF   41975812ull
#define RST_OFF   41975813ull

// d_ws layout (bytes):
//   [0..32)        float accumulators: 0=var_sum 1=min2_sum 2=hsw_sum 3=rst_sum 4=kl_sum
//   [64..4160)     float norms[NE]                  (float offset 16)
//   [4224..135296) int ind[B_ROWS]                  (float offset 1056)
//   [256K..768K)   __bf16 pw_bf[NE*ED]
//   [768K..1280K)  __bf16 cb_bf[NE*ED]
#define WS_PWBF_BYTES  (262144u)
#define WS_CBBF_BYTES  (786432u)

__device__ __forceinline__ __bf16 f2bf(float f) { return (__bf16)f; }  // hw v_cvt path

// Concatenate two aligned 16B bf16 loads into one WMMA fragment register set.
__device__ __forceinline__ v16bf load_frag_bf(const __bf16* __restrict__ base,
                                              int kb, int h8) {
  v8bf lo = *reinterpret_cast<const v8bf*>(base + kb + h8);        // K = kb+h8 .. +7
  v8bf hi = *reinterpret_cast<const v8bf*>(base + kb + 16 + h8);   // K = kb+16+h8 .. +7
  return __builtin_shufflevector(lo, hi, 0, 1, 2, 3, 4, 5, 6, 7,
                                 8, 9, 10, 11, 12, 13, 14, 15);
}

// Load a full 16xK B tile (8 fragments) — issues as one load clause.
__device__ __forceinline__ void load_btile(v16bf* __restrict__ b,
                                           const __bf16* __restrict__ base, int h8) {
#pragma unroll
  for (int ks = 0; ks < 8; ++ks) b[ks] = load_frag_bf(base, ks * 32, h8);
}

// 8-deep WMMA chain for one 16x16 output tile (K = 256).
__device__ __forceinline__ v8f wmma_tile(const v16bf* __restrict__ a,
                                         const v16bf* __restrict__ b) {
  v8f c = {};
#pragma unroll
  for (int ks = 0; ks < 8; ++ks)
    c = __builtin_amdgcn_wmma_f32_16x16x32_bf16(false, a[ks], false, b[ks],
                                                (short)0, c, false, false);
  return c;
}

// Async global->LDS copy of one 4 KB gumbel row: 8 x (32 lanes x b128).
// ISA 15.18.3 GV mode: global_load_async_to_lds_b128 vdst(LDS addr), vaddr64, off
__device__ __forceinline__ void async_row_to_lds(const float* __restrict__ gsrc,
                                                 unsigned lds_base, int lane) {
#pragma unroll
  for (int ch = 0; ch < 8; ++ch) {
    unsigned lds_off = lds_base + (unsigned)(ch * 512 + lane * 16);
    unsigned long long ga =
        (unsigned long long)(const void*)(gsrc + ch * 128 + lane * 4);
    asm volatile("global_load_async_to_lds_b128 %0, %1, off"
                 :: "v"(lds_off), "v"(ga) : "memory");
  }
}

// ---------------------------------------------------------------------------
// Zero the sampled_idx output region and the scalar accumulators.
__global__ void init_kernel(float* __restrict__ out, float* __restrict__ wsf) {
  size_t idx = (size_t)blockIdx.x * blockDim.x + threadIdx.x;
  if (idx < 8) wsf[idx] = 0.0f;
  size_t stride = (size_t)gridDim.x * blockDim.x;
  for (size_t i = idx; i < (size_t)B_ROWS * NE; i += stride)
    out[SAMP_OFF + i] = 0.0f;
}

// ---------------------------------------------------------------------------
// fp32 -> bf16 packing for the L2-resident GEMM operands (proj_w, cb).
__global__ void pack_bf16_kernel(const float* __restrict__ src,
                                 __bf16* __restrict__ dst, int n) {
  int i = blockIdx.x * blockDim.x + threadIdx.x;
  int stride = gridDim.x * blockDim.x;
  for (; i < n; i += stride) dst[i] = f2bf(src[i]);
}

// ---------------------------------------------------------------------------
// Per-codebook-row norms, hypersphere loss partial, clamp(r) mean partial.
__global__ void norms_kernel(const float* __restrict__ cb, const float* __restrict__ r,
                             float* __restrict__ wsf) {
  int wave = threadIdx.x >> 5, lane = threadIdx.x & 31;
  int row  = blockIdx.x * 8 + wave;
  if (row >= NE) return;
  const float* cr = cb + (size_t)row * ED;
  float s = 0.0f;
#pragma unroll
  for (int i = 0; i < 8; ++i) { float v = cr[lane + 32 * i]; s += v * v; }
#pragma unroll
  for (int off = 16; off; off >>= 1) s += __shfl_xor(s, off, 32);
  if (lane == 0) {
    float nrm = sqrtf(s);
    wsf[16 + row] = nrm;
    float d = r[row] - nrm;
    atomicAdd(&wsf[2], d * d);
    atomicAdd(&wsf[3], fminf(fmaxf(r[row], 0.9f), 1.1f));
  }
}

// ---------------------------------------------------------------------------
// Codebook geodesic-distance stats: one wave owns rows [i0,i0+16), sweeps all
// 1024 columns with v_wmma_f32_16x16x32_bf16. d1 is symmetric, so the
// column-wise 2nd-smallest equals the row-wise 2nd-smallest.
__global__ __launch_bounds__(32) void cbstats_kernel(const __bf16* __restrict__ cbbf,
                                                     const float* __restrict__ norms,
                                                     float* __restrict__ wsf) {
  __shared__ float tile[16][17];
  int lane = threadIdx.x & 31;
  int l = lane & 15, h = lane >> 4, h8 = h * 8;
  int i0 = blockIdx.x * 16;

  const __bf16* ar = cbbf + (size_t)(i0 + l) * ED;
  v16bf afrag[8];
  load_btile(afrag, ar, h8);

  float rinv_i[8];
#pragma unroll
  for (int v = 0; v < 8; ++v) rinv_i[v] = __builtin_amdgcn_rcpf(norms[i0 + v + 8 * h]);

  float rsum = 0.f, rsumsq = 0.f, rmin1 = 1e30f, rmin2 = 1e30f;

  for (int jt = 0; jt < NE / 16; ++jt) {
    int j0 = jt * 16;
    v16bf b[8];
    load_btile(b, cbbf + (size_t)(j0 + l) * ED, h8);   // B[k][n] = cb[n][k]
    v8f c = wmma_tile(afrag, b);
    float rnj = __builtin_amdgcn_rcpf(norms[j0 + l]);
#pragma unroll
    for (int v = 0; v < 8; ++v) {
      float ca = c[v] * rinv_i[v] * rnj;
      ca = fminf(fmaxf(ca, -0.99999f), 0.99999f);
      tile[v + 8 * h][l] = acosf(ca);
    }
    __syncthreads();
    if (lane < 16) {
#pragma unroll
      for (int jj = 0; jj < 16; ++jj) {
        float d = tile[lane][jj];
        rsum += d; rsumsq += d * d;
        if (d < rmin1) { rmin2 = rmin1; rmin1 = d; }
        else if (d < rmin2) { rmin2 = d; }
      }
    }
    __syncthreads();
  }
  if (lane < 16) {
    float var = (rsumsq - rsum * rsum * (1.0f / NE)) * (1.0f / (NE - 1));
    atomicAdd(&wsf[0], var);    // sum of per-row variances
    atomicAdd(&wsf[1], rmin2);  // sum of 2nd-smallest distances
  }
}

// ---------------------------------------------------------------------------
// Fused logits GEMM (WMMA bf16, fp32 accum) + Gumbel argmax + online softmax
// KL + z_q gather. One wave owns a 16-row slab; logits live only in LDS.
// B fragments are double-buffered in registers; gumbel rows are double-
// buffered in LDS via global_load_async_to_lds_b128 (row 0 issued before the
// GEMM so its HBM latency hides behind the 512-WMMA chain).
__global__ __launch_bounds__(32) void logits_kernel(
    const float* __restrict__ z, const float* __restrict__ cb,
    const __bf16* __restrict__ pwbf, const float* __restrict__ pb,
    const float* __restrict__ gumbel, float* __restrict__ out,
    float* __restrict__ wsf, int* __restrict__ wsind) {
  __shared__ float lg[16][NE];                 // 64 KB logits slab
  __shared__ float gbuf[2][NE];                // 2 x 4 KB gumbel row buffers
  int lane = threadIdx.x & 31;
  int l = lane & 15, h = lane >> 4, h8 = h * 8;
  int r0 = blockIdx.x * 16;

  unsigned gb_lds[2];
  gb_lds[0] = (unsigned)(size_t)(&gbuf[0][0]);
  gb_lds[1] = (unsigned)(size_t)(&gbuf[1][0]);
  // Kick off the async fetch of gumbel row 0 before the GEMM phase.
  async_row_to_lds(gumbel + (size_t)r0 * NE, gb_lds[0], lane);

  // A fragments: convert z row once (amortized over 64 N-tiles).
  const float* zr = z + (size_t)(r0 + l) * ED;
  v16bf afrag[8];
#pragma unroll
  for (int ks = 0; ks < 8; ++ks) {
    int kb = ks * 32;
#pragma unroll
    for (int i = 0; i < 8; ++i) afrag[ks][i]     = f2bf(zr[kb + h8 + i]);
#pragma unroll
    for (int i = 0; i < 8; ++i) afrag[ks][8 + i] = f2bf(zr[kb + 16 + h8 + i]);
  }

  v16bf b0[8], b1[8];
  load_btile(b0, pwbf + (size_t)l * ED, h8);   // tile 0
  for (int nt = 0; nt < NE / 16; nt += 2) {
    load_btile(b1, pwbf + (size_t)((nt + 1) * 16 + l) * ED, h8);
    {
      v8f c = wmma_tile(afrag, b0);
      int n0 = nt * 16;
      float bias = pb[n0 + l];
#pragma unroll
      for (int v = 0; v < 8; ++v) lg[v + 8 * h][n0 + l] = c[v] + bias;
    }
    if (nt + 2 < NE / 16)
      load_btile(b0, pwbf + (size_t)((nt + 2) * 16 + l) * ED, h8);
    {
      v8f c = wmma_tile(afrag, b1);
      int n0 = nt * 16 + 16;
      float bias = pb[n0 + l];
#pragma unroll
      for (int v = 0; v < 8; ++v) lg[v + 8 * h][n0 + l] = c[v] + bias;
    }
  }
  __syncthreads();

  for (int row = 0; row < 16; ++row) {
    int grow = r0 + row;
    // Prefetch next gumbel row, then drain the current one (ASYNCcnt retires
    // in order: wait<=8 leaves only the 8 just-issued ops outstanding).
    if (row + 1 < 16) {
      async_row_to_lds(gumbel + (size_t)(grow + 1) * NE, gb_lds[(row + 1) & 1], lane);
      asm volatile("s_wait_asynccnt 0x8" ::: "memory");
    } else {
      asm volatile("s_wait_asynccnt 0x0" ::: "memory");
    }
    const float* gr = &gbuf[row & 1][0];

    float maxg = -1e30f, maxq = -1e30f; int argn = 0;
#pragma unroll 4
    for (int t = 0; t < 32; ++t) {
      int n = lane + t * 32;
      float lq = lg[row][n];
      float v  = lq + gr[n];                 // TAU == 1
      if (v > maxg) { maxg = v; argn = n; }
      maxq = fmaxf(maxq, lq);
    }
#pragma unroll
    for (int off = 16; off; off >>= 1) {
      float om = __shfl_xor(maxg, off, 32);
      int   oi = __shfl_xor(argn, off, 32);
      if (om > maxg || (om == maxg && oi < argn)) { maxg = om; argn = oi; }
      maxq = fmaxf(maxq, __shfl_xor(maxq, off, 32));
    }
    float S0 = 0.f, S1 = 0.f;
#pragma unroll 4
    for (int t = 0; t < 32; ++t) {
      float x = lg[row][lane + t * 32] - maxq;
      float e = __expf(x);
      S0 += e; S1 += e * x;
    }
#pragma unroll
    for (int off = 16; off; off >>= 1) {
      S0 += __shfl_xor(S0, off, 32);
      S1 += __shfl_xor(S1, off, 32);
    }
    if (lane == 0) {
      // sum_j qy*log(qy*n_e) = S1/S0 - log(S0) + log(NE)
      float klrow = S1 / S0 - __logf(S0) + 6.93147180559945f;
      atomicAdd(&wsf[4], klrow);
      wsind[grow] = argn;
      out[IND_OFF + grow] = (float)argn;
    }
    // z_q[grow,:] = cb[argn,:]   (y_hard gather; argn identical on all lanes)
    const float* cr = cb + (size_t)argn * ED;
    float* zq = out + ZQ_OFF + (size_t)grow * ED;
#pragma unroll
    for (int k = lane; k < ED; k += 32) zq[k] = cr[k];
  }
}

// ---------------------------------------------------------------------------
__global__ void scatter_kernel(const int* __restrict__ wsind, float* __restrict__ out) {
  int b = blockIdx.x * blockDim.x + threadIdx.x;
  if (b < B_ROWS) out[SAMP_OFF + (size_t)wsind[b]] = 1.0f;  // flat scatter, idx < NE
}

__global__ void finalize_kernel(const float* __restrict__ wsf, float* __restrict__ out) {
  if (threadIdx.x == 0 && blockIdx.x == 0) {
    float cbvar = wsf[0] * (1.0f / NE);
    float mind  = wsf[1] * (1.0f / NE);
    float hsw   = wsf[2] * (1.0f / NE);
    float rstm  = wsf[3] * (1.0f / NE);
    float kl    = KL_W * (wsf[4] * (1.0f / B_ROWS));
    out[CBV_OFF]  = cbvar;
    out[MIND_OFF] = mind;
    out[HSW_OFF]  = hsw;
    out[CBL_OFF]  = 0.0f;
    out[RST_OFF]  = rstm;
    out[LOSS_OFF] = kl + hsw + cbvar - mind;
  }
}

// ---------------------------------------------------------------------------
extern "C" void kernel_launch(void* const* d_in, const int* in_sizes, int n_in,
                              void* d_out, int out_size, void* d_ws, size_t ws_size,
                              hipStream_t stream) {
  const float* z   = (const float*)d_in[0];
  const float* cb  = (const float*)d_in[1];
  const float* pw  = (const float*)d_in[2];
  const float* pb  = (const float*)d_in[3];
  const float* r   = (const float*)d_in[4];
  const float* gum = (const float*)d_in[5];
  float*  out   = (float*)d_out;
  float*  wsf   = (float*)d_ws;
  int*    wsind = (int*)((char*)d_ws + 1056 * sizeof(float));
  __bf16* pwbf  = (__bf16*)((char*)d_ws + WS_PWBF_BYTES);
  __bf16* cbbf  = (__bf16*)((char*)d_ws + WS_CBBF_BYTES);

  init_kernel     <<<2048, 256, 0, stream>>>(out, wsf);
  pack_bf16_kernel<<<256, 256, 0, stream>>>(pw, pwbf, NE * ED);
  pack_bf16_kernel<<<256, 256, 0, stream>>>(cb, cbbf, NE * ED);
  norms_kernel    <<<NE / 8, 256, 0, stream>>>(cb, r, wsf);
  cbstats_kernel  <<<NE / 16, 32, 0, stream>>>(cbbf, wsf + 16, wsf);
  logits_kernel   <<<B_ROWS / 16, 32, 0, stream>>>(z, cb, pwbf, pb, gum, out, wsf, wsind);
  scatter_kernel  <<<B_ROWS / 256, 256, 0, stream>>>(wsind, out);
  finalize_kernel <<<1, 32, 0, stream>>>(wsf, out);
}
